// PolyHashV3_29910152249489
// MI455X (gfx1250) — matrix operand order repaired and checked by reading, not compiled, not run on hardware
//
#include <hip/hip_runtime.h>
#include <hip/hip_bf16.h>

typedef __bf16 bf16;
typedef __attribute__((ext_vector_type(16))) __bf16 v16bf;
typedef __attribute__((ext_vector_type(8)))  __bf16 v8bf;
typedef __attribute__((ext_vector_type(8)))  float  v8f;

#define T_SEQ 512
#define NROWS 32768   // 64 * 512 tokens
#define KF    448     // 426 features padded to multiple of 32
#define HID   512
#define VOUT  1024
#define LDP   40      // LDS tile row pitch in bf16 elements (80B, 16B-aligned, bank-spread)

__device__ __constant__ unsigned int cPRIMES[32] = {
  2654435761u, 2246822519u, 3266489917u, 2028178513u, 1220703125u, 1610612741u,
  805306457u, 402653189u, 3674653429u, 2860486313u, 1073676287u, 2971215073u,
  1500450271u, 3267000013u, 2654435789u, 4049292737u, 2246822531u, 3266489927u,
  2028178519u, 1220703133u, 1610612743u, 805306459u, 402653191u, 3674653433u,
  2654435771u, 2246822527u, 3266489933u, 2028178529u, 1220703137u, 1610612747u,
  805306463u, 402653197u};
__device__ __constant__ int cFIB[8]   = {1, 1, 2, 3, 5, 8, 13, 21};
__device__ __constant__ int cMATCH[14]= {1, 2, 3, 4, 5, 6, 7, 8, 10, 12, 16, 20, 24, 32};
__device__ __constant__ int cDELTA[5] = {1, 2, 3, 5, 8};
__device__ __constant__ int cFREQW[4] = {4, 8, 16, 32};

// ---------------- fingerprint scan: fp = (fp >> sr) ^ (tok * prime) ----------
__global__ void phv3_fp_scan(const int* __restrict__ chars, int* __restrict__ fpb) {
  int tid = blockIdx.x * blockDim.x + threadIdx.x;
  if (tid >= 3 * 64) return;
  int fi = tid / 64, b = tid % 64;
  const int SR[3] = {1, 2, 4};
  unsigned long long prime = cPRIMES[fi];
  int sr = SR[fi];
  unsigned long long fp = 0ull;
  const int* cp = chars + b * T_SEQ;
  int* op = fpb + fi * NROWS + b * T_SEQ;
  for (int t = 0; t < T_SEQ; ++t) {
    fp = (fp >> sr) ^ ((unsigned long long)(unsigned int)cp[t] * prime);
    op[t] = (int)(fp & 16383ull);   // abs(fp % 16384), fp >= 0
  }
}

// ---------------- feature builder: wave per token, lane per column ----------
__global__ void __launch_bounds__(256)
phv3_features(const int* __restrict__ chars, const float* __restrict__ byte_embed,
              const float* __restrict__ hash_tables, const float* __restrict__ delta_tables,
              const float* __restrict__ fp_tables, const int* __restrict__ fpb,
              bf16* __restrict__ F) {
  int row  = blockIdx.x * 8 + (threadIdx.x >> 5);
  int lane = threadIdx.x & 31;
  int b = row >> 9;
  int t = row & 511;
  const int* cp = chars + b * T_SEQ;
  int c = cp[t];
  bf16* out = F + (size_t)row * KF;
  for (int f = lane; f < KF; f += 32) {
    float val = 0.f;
    if (f < 128) {
      val = byte_embed[c * 128 + f];
    } else if (f < 256) {                       // 8 polynomial-hash tables
      int i = (f - 128) >> 4, e = (f - 128) & 15;
      int off = cFIB[i];
      int sc = (t >= off) ? cp[t - off] : 0;
      unsigned idx = ((unsigned)sc * cPRIMES[(3 * i) & 31]) & 32767u;
      val = hash_tables[((size_t)i * 32768 + idx) * 16 + e];
    } else if (f < 270) {                       // match features
      int k = cMATCH[f - 256];
      int sc = (t >= k) ? cp[t - k] : 0;
      val = (t >= k && c == sc) ? 1.f : 0.f;
    } else if (f < 310) {                       // delta-hash tables
      int i = (f - 270) >> 3, e = (f - 270) & 7;
      int k = cDELTA[i];
      int sc = (t >= k) ? cp[t - k] : 0;
      unsigned d = (unsigned)(c - sc) & 1023u;  // mod V (power of two)
      unsigned bucket = (d * cPRIMES[i]) & 2047u;
      val = delta_tables[((size_t)i * 2048 + bucket) * 8 + e];
    } else if (f < 314) {                       // local frequency
      int w = cFREQW[f - 310];
      int cnt = 0;
      for (int lag = 1; lag <= w; ++lag) {
        int sc = (t >= lag) ? cp[t - lag] : 0;  // zero-pad semantics (unmasked)
        cnt += (c == sc) ? 1 : 0;
      }
      val = (float)cnt / (float)w;
    } else if (f < 362) {                       // fingerprint tables
      int i = (f - 314) >> 4, e = (f - 314) & 15;
      int bucket = fpb[i * NROWS + row];
      val = fp_tables[((size_t)i * 16384 + bucket) * 16 + e];
    } else if (f < 426) {                       // count-min sketch
      int j = f - 362;
      int s = 0;
      for (int lag = 1; lag <= 8; ++lag) {
        int sc = (t >= lag) ? cp[t - lag] : 0;
        unsigned bin = ((unsigned)sc * cPRIMES[lag]) & 63u;
        s += (bin == (unsigned)j) ? 1 : 0;
      }
      val = (float)s * 0.125f;
    }
    out[f] = (bf16)val;                         // f >= 426 -> zero pad
  }
}

// ---------------- f32 -> bf16 weight conversion with K padding -------------
__global__ void phv3_cvt(const float* __restrict__ src, bf16* __restrict__ dst,
                         int rows, int K, int ldd) {
  int i = blockIdx.x * blockDim.x + threadIdx.x;
  if (i >= rows * ldd) return;
  int r = i / ldd, col = i - r * ldd;
  dst[i] = (bf16)((col < K) ? src[r * K + col] : 0.f);
}

// ---------------- async global -> LDS staging (gfx1250 ASYNCcnt path) -------
__device__ __forceinline__ void phv3_async_b128(bf16* lds_ptr, const bf16* gptr) {
  unsigned loff = (unsigned)(unsigned long long)lds_ptr;   // flat LDS addr[31:0] = group offset
  unsigned long long ga = (unsigned long long)gptr;
  asm volatile("global_load_async_to_lds_b128 %0, %1, off"
               :: "v"(loff), "v"(ga) : "memory");
}
__device__ __forceinline__ void phv3_wait_async0() {
  asm volatile("s_wait_asynccnt 0x0" ::: "memory");
}

// Stage a 128x32 A tile (rows of A, one k-step) into LDS: 512 16B chunks / 256 thr
__device__ __forceinline__ void phv3_issue_tileA(const bf16* __restrict__ A, int lda,
                                                 int m0, int k0, bf16* sA, int tid) {
  #pragma unroll
  for (int c = 0; c < 2; ++c) {
    int chunk = tid + c * 256;
    int row = chunk >> 2, col = (chunk & 3) << 3;   // 4 x 8-elem chunks per row
    phv3_async_b128(sA + row * LDP + col, A + (size_t)(m0 + row) * lda + k0 + col);
  }
}
// Stage a 64x32 B tile: 256 16B chunks / 256 thr
__device__ __forceinline__ void phv3_issue_tileB(const bf16* __restrict__ W, int ldb,
                                                 int n0, int k0, bf16* sB, int tid) {
  int row = tid >> 2, col = (tid & 3) << 3;
  phv3_async_b128(sB + row * LDP + col, W + (size_t)(n0 + row) * ldb + k0 + col);
}

// ---------------- WMMA fragment helpers (from LDS tiles) --------------------
union AB16 { v16bf v; v8bf h[2]; };

// A 16x32 bf16 fragment: lane L = row (L&15); K = {k+8h..k+8h+7, k+16+8h..}
__device__ __forceinline__ v16bf phv3_lds_a(const bf16* sA, int mloc, int lane) {
  const bf16* p = sA + (mloc + (lane & 15)) * LDP + ((lane >> 4) << 3);
  AB16 r;
  r.h[0] = *(const v8bf*)(p);
  r.h[1] = *(const v8bf*)(p + 16);
  return r.v;
}
// B 32x16 fragment from row-major W[N][K] tile: lane L = col (L&15); K = {k+8h..k+8h+7, k+16+8h..}
__device__ __forceinline__ v16bf phv3_lds_b(const bf16* sB, int nloc, int lane) {
  const bf16* p = sB + (nloc + (lane & 15)) * LDP + ((lane >> 4) << 3);
  AB16 r;
  r.h[0] = *(const v8bf*)(p);
  r.h[1] = *(const v8bf*)(p + 16);
  return r.v;
}

__device__ __forceinline__ v8f phv3_wmma(v16bf a, v16bf b, v8f c) {
  return __builtin_amdgcn_wmma_f32_16x16x32_bf16(false, a, false, b, (short)0, c,
                                                 false, false);
}

__device__ __forceinline__ float phv3_silu(float x) { return x / (1.f + __expf(-x)); }

// ---------------- GEMM1: H = silu(F @ in_W^T + in_b) -----------------------
__global__ void __launch_bounds__(256)
phv3_gemm1(const bf16* __restrict__ A, const bf16* __restrict__ W,
           const float* __restrict__ bias, bf16* __restrict__ H) {
  __shared__ bf16 sA[2][128 * LDP];
  __shared__ bf16 sB[2][64 * LDP];
  int tid = threadIdx.x, lane = tid & 31, w = tid >> 5;
  int m0 = blockIdx.x * 128 + (w >> 1) * 32;
  int n0 = blockIdx.y * 64 + (w & 1) * 32;
  int mloc = (w >> 1) * 32, nloc = (w & 1) * 32;
  v8f cc[2][2] = {};
  phv3_issue_tileA(A, KF, blockIdx.x * 128, 0, sA[0], tid);
  phv3_issue_tileB(W, KF, blockIdx.y * 64, 0, sB[0], tid);
  const int KT = KF / 32;
  for (int kk = 0; kk < KT; ++kk) {
    int p = kk & 1;
    phv3_wait_async0();
    __syncthreads();
    if (kk + 1 < KT) {
      phv3_issue_tileA(A, KF, blockIdx.x * 128, (kk + 1) * 32, sA[p ^ 1], tid);
      phv3_issue_tileB(W, KF, blockIdx.y * 64, (kk + 1) * 32, sB[p ^ 1], tid);
    }
    v16bf a0 = phv3_lds_a(sA[p], mloc, lane);
    v16bf a1 = phv3_lds_a(sA[p], mloc + 16, lane);
    v16bf b0 = phv3_lds_b(sB[p], nloc, lane);
    v16bf b1 = phv3_lds_b(sB[p], nloc + 16, lane);
    cc[0][0] = phv3_wmma(a0, b0, cc[0][0]);
    cc[0][1] = phv3_wmma(a0, b1, cc[0][1]);
    cc[1][0] = phv3_wmma(a1, b0, cc[1][0]);
    cc[1][1] = phv3_wmma(a1, b1, cc[1][1]);
  }
  int mrow = (lane >> 4) * 8, ncol = lane & 15;
  #pragma unroll
  for (int i = 0; i < 2; ++i)
    #pragma unroll
    for (int j = 0; j < 2; ++j)
      #pragma unroll
      for (int r = 0; r < 8; ++r) {
        int m = m0 + i * 16 + mrow + r, n = n0 + j * 16 + ncol;
        float x = cc[i][j][r] + bias[n];
        H[(size_t)m * HID + n] = (bf16)phv3_silu(x);
      }
}

// ---------------- GEMM2: U = silu(H @ w1^T) * (H @ w2^T) -------------------
__global__ void __launch_bounds__(256)
phv3_gemm2(const bf16* __restrict__ A, const bf16* __restrict__ W1,
           const bf16* __restrict__ W2, bf16* __restrict__ U) {
  __shared__ bf16 sA[2][128 * LDP];
  __shared__ bf16 sB1[2][64 * LDP];
  __shared__ bf16 sB2[2][64 * LDP];
  int tid = threadIdx.x, lane = tid & 31, w = tid >> 5;
  int m0 = blockIdx.x * 128 + (w >> 1) * 32;
  int n0 = blockIdx.y * 64 + (w & 1) * 32;
  int mloc = (w >> 1) * 32, nloc = (w & 1) * 32;
  v8f c1[2][2] = {}, c2[2][2] = {};
  phv3_issue_tileA(A, HID, blockIdx.x * 128, 0, sA[0], tid);
  phv3_issue_tileB(W1, HID, blockIdx.y * 64, 0, sB1[0], tid);
  phv3_issue_tileB(W2, HID, blockIdx.y * 64, 0, sB2[0], tid);
  const int KT = HID / 32;
  for (int kk = 0; kk < KT; ++kk) {
    int p = kk & 1;
    phv3_wait_async0();
    __syncthreads();
    if (kk + 1 < KT) {
      phv3_issue_tileA(A, HID, blockIdx.x * 128, (kk + 1) * 32, sA[p ^ 1], tid);
      phv3_issue_tileB(W1, HID, blockIdx.y * 64, (kk + 1) * 32, sB1[p ^ 1], tid);
      phv3_issue_tileB(W2, HID, blockIdx.y * 64, (kk + 1) * 32, sB2[p ^ 1], tid);
    }
    v16bf a0 = phv3_lds_a(sA[p], mloc, lane);
    v16bf a1 = phv3_lds_a(sA[p], mloc + 16, lane);
    v16bf p0 = phv3_lds_b(sB1[p], nloc, lane);
    v16bf p1 = phv3_lds_b(sB1[p], nloc + 16, lane);
    v16bf q0 = phv3_lds_b(sB2[p], nloc, lane);
    v16bf q1 = phv3_lds_b(sB2[p], nloc + 16, lane);
    c1[0][0] = phv3_wmma(a0, p0, c1[0][0]);
    c1[0][1] = phv3_wmma(a0, p1, c1[0][1]);
    c1[1][0] = phv3_wmma(a1, p0, c1[1][0]);
    c1[1][1] = phv3_wmma(a1, p1, c1[1][1]);
    c2[0][0] = phv3_wmma(a0, q0, c2[0][0]);
    c2[0][1] = phv3_wmma(a0, q1, c2[0][1]);
    c2[1][0] = phv3_wmma(a1, q0, c2[1][0]);
    c2[1][1] = phv3_wmma(a1, q1, c2[1][1]);
  }
  int mrow = (lane >> 4) * 8, ncol = lane & 15;
  #pragma unroll
  for (int i = 0; i < 2; ++i)
    #pragma unroll
    for (int j = 0; j < 2; ++j)
      #pragma unroll
      for (int r = 0; r < 8; ++r) {
        int m = m0 + i * 16 + mrow + r, n = n0 + j * 16 + ncol;
        U[(size_t)m * HID + n] = (bf16)(phv3_silu(c1[i][j][r]) * c2[i][j][r]);
      }
}

// ---------------- GEMM3: X = U @ w3^T + H ----------------------------------
__global__ void __launch_bounds__(256)
phv3_gemm3(const bf16* __restrict__ A, const bf16* __restrict__ W,
           const bf16* __restrict__ H, bf16* __restrict__ X) {
  __shared__ bf16 sA[2][128 * LDP];
  __shared__ bf16 sB[2][64 * LDP];
  int tid = threadIdx.x, lane = tid & 31, w = tid >> 5;
  int m0 = blockIdx.x * 128 + (w >> 1) * 32;
  int n0 = blockIdx.y * 64 + (w & 1) * 32;
  int mloc = (w >> 1) * 32, nloc = (w & 1) * 32;
  v8f cc[2][2] = {};
  phv3_issue_tileA(A, HID, blockIdx.x * 128, 0, sA[0], tid);
  phv3_issue_tileB(W, HID, blockIdx.y * 64, 0, sB[0], tid);
  const int KT = HID / 32;
  for (int kk = 0; kk < KT; ++kk) {
    int p = kk & 1;
    phv3_wait_async0();
    __syncthreads();
    if (kk + 1 < KT) {
      phv3_issue_tileA(A, HID, blockIdx.x * 128, (kk + 1) * 32, sA[p ^ 1], tid);
      phv3_issue_tileB(W, HID, blockIdx.y * 64, (kk + 1) * 32, sB[p ^ 1], tid);
    }
    v16bf a0 = phv3_lds_a(sA[p], mloc, lane);
    v16bf a1 = phv3_lds_a(sA[p], mloc + 16, lane);
    v16bf b0 = phv3_lds_b(sB[p], nloc, lane);
    v16bf b1 = phv3_lds_b(sB[p], nloc + 16, lane);
    cc[0][0] = phv3_wmma(a0, b0, cc[0][0]);
    cc[0][1] = phv3_wmma(a0, b1, cc[0][1]);
    cc[1][0] = phv3_wmma(a1, b0, cc[1][0]);
    cc[1][1] = phv3_wmma(a1, b1, cc[1][1]);
  }
  int mrow = (lane >> 4) * 8, ncol = lane & 15;
  #pragma unroll
  for (int i = 0; i < 2; ++i)
    #pragma unroll
    for (int j = 0; j < 2; ++j)
      #pragma unroll
      for (int r = 0; r < 8; ++r) {
        int m = m0 + i * 16 + mrow + r, n = n0 + j * 16 + ncol;
        float x = cc[i][j][r] + (float)H[(size_t)m * HID + n];
        X[(size_t)m * HID + n] = (bf16)x;
      }
}

// ---------------- LayerNorm: wave per row (512 cols, 16 per lane) ----------
__global__ void __launch_bounds__(256)
phv3_ln(const bf16* __restrict__ X, const float* __restrict__ g,
        const float* __restrict__ bb, bf16* __restrict__ Xn) {
  int row = blockIdx.x * 8 + (threadIdx.x >> 5);
  int lane = threadIdx.x & 31;
  const bf16* xp = X + (size_t)row * HID;
  float v[16], s = 0.f, s2 = 0.f;
  #pragma unroll
  for (int i = 0; i < 16; ++i) {
    float x = (float)xp[lane + i * 32];
    v[i] = x; s += x; s2 += x * x;
  }
  #pragma unroll
  for (int m = 16; m >= 1; m >>= 1) {
    s  += __shfl_xor(s, m, 32);
    s2 += __shfl_xor(s2, m, 32);
  }
  float mu = s * (1.f / 512.f);
  float var = s2 * (1.f / 512.f) - mu * mu;
  float inv = rsqrtf(var + 1e-5f);
  bf16* op = Xn + (size_t)row * HID;
  #pragma unroll
  for (int i = 0; i < 16; ++i) {
    int col = lane + i * 32;
    op[col] = (bf16)((v[i] - mu) * inv * g[col] + bb[col]);
  }
}

// ---------------- GEMM4: out = Xn @ out_W^T + out_b (f32 out) --------------
__global__ void __launch_bounds__(256)
phv3_gemm4(const bf16* __restrict__ A, const bf16* __restrict__ W,
           const float* __restrict__ bias, float* __restrict__ out) {
  __shared__ bf16 sA[2][128 * LDP];
  __shared__ bf16 sB[2][64 * LDP];
  int tid = threadIdx.x, lane = tid & 31, w = tid >> 5;
  int m0 = blockIdx.x * 128 + (w >> 1) * 32;
  int n0 = blockIdx.y * 64 + (w & 1) * 32;
  int mloc = (w >> 1) * 32, nloc = (w & 1) * 32;
  v8f cc[2][2] = {};
  phv3_issue_tileA(A, HID, blockIdx.x * 128, 0, sA[0], tid);
  phv3_issue_tileB(W, HID, blockIdx.y * 64, 0, sB[0], tid);
  const int KT = HID / 32;
  for (int kk = 0; kk < KT; ++kk) {
    int p = kk & 1;
    phv3_wait_async0();
    __syncthreads();
    if (kk + 1 < KT) {
      phv3_issue_tileA(A, HID, blockIdx.x * 128, (kk + 1) * 32, sA[p ^ 1], tid);
      phv3_issue_tileB(W, HID, blockIdx.y * 64, (kk + 1) * 32, sB[p ^ 1], tid);
    }
    v16bf a0 = phv3_lds_a(sA[p], mloc, lane);
    v16bf a1 = phv3_lds_a(sA[p], mloc + 16, lane);
    v16bf b0 = phv3_lds_b(sB[p], nloc, lane);
    v16bf b1 = phv3_lds_b(sB[p], nloc + 16, lane);
    cc[0][0] = phv3_wmma(a0, b0, cc[0][0]);
    cc[0][1] = phv3_wmma(a0, b1, cc[0][1]);
    cc[1][0] = phv3_wmma(a1, b0, cc[1][0]);
    cc[1][1] = phv3_wmma(a1, b1, cc[1][1]);
  }
  int mrow = (lane >> 4) * 8, ncol = lane & 15;
  #pragma unroll
  for (int i = 0; i < 2; ++i)
    #pragma unroll
    for (int j = 0; j < 2; ++j)
      #pragma unroll
      for (int r = 0; r < 8; ++r) {
        int m = m0 + i * 16 + mrow + r, n = n0 + j * 16 + ncol;
        out[(size_t)m * VOUT + n] = cc[i][j][r] + bias[n];
      }
}

// ---------------- launch ----------------------------------------------------
extern "C" void kernel_launch(void* const* d_in, const int* in_sizes, int n_in,
                              void* d_out, int out_size, void* d_ws, size_t ws_size,
                              hipStream_t stream) {
  (void)in_sizes; (void)n_in; (void)out_size; (void)ws_size;
  const int*   chars        = (const int*)  d_in[0];
  const float* byte_embed   = (const float*)d_in[1];
  const float* hash_tables  = (const float*)d_in[2];
  const float* delta_tables = (const float*)d_in[3];
  const float* fp_tables    = (const float*)d_in[4];
  const float* in_W  = (const float*)d_in[5];
  const float* in_b  = (const float*)d_in[6];
  const float* w1    = (const float*)d_in[7];
  const float* w2    = (const float*)d_in[8];
  const float* w3    = (const float*)d_in[9];
  const float* ln_g  = (const float*)d_in[10];
  const float* ln_b  = (const float*)d_in[11];
  const float* out_W = (const float*)d_in[12];
  const float* out_b = (const float*)d_in[13];

  char* ws = (char*)d_ws;
  size_t off = 0;
  auto carve = [&](size_t bytes) -> void* {
    off = (off + 255) & ~(size_t)255;
    void* p = ws + off;
    off += bytes;
    return p;
  };
  int*  fpb = (int*) carve((size_t)3 * NROWS * 4);
  bf16* F   = (bf16*)carve((size_t)NROWS * KF * 2);
  bf16* Wa  = (bf16*)carve((size_t)HID * KF * 2);
  bf16* W1b = (bf16*)carve((size_t)HID * HID * 2);
  bf16* W2b = (bf16*)carve((size_t)HID * HID * 2);
  bf16* W3b = (bf16*)carve((size_t)HID * HID * 2);
  bf16* Wob = (bf16*)carve((size_t)VOUT * HID * 2);
  bf16* H   = (bf16*)carve((size_t)NROWS * HID * 2);
  bf16* U   = (bf16*)carve((size_t)NROWS * HID * 2);
  bf16* X   = (bf16*)carve((size_t)NROWS * HID * 2);
  bf16* Xn  = (bf16*)carve((size_t)NROWS * HID * 2);

  phv3_fp_scan<<<1, 256, 0, stream>>>(chars, fpb);
  phv3_features<<<NROWS / 8, 256, 0, stream>>>(chars, byte_embed, hash_tables,
                                               delta_tables, fp_tables, fpb, F);
  phv3_cvt<<<(HID * KF + 255) / 256, 256, 0, stream>>>(in_W, Wa, HID, 426, KF);
  phv3_cvt<<<(HID * HID + 255) / 256, 256, 0, stream>>>(w1, W1b, HID, HID, HID);
  phv3_cvt<<<(HID * HID + 255) / 256, 256, 0, stream>>>(w2, W2b, HID, HID, HID);
  phv3_cvt<<<(HID * HID + 255) / 256, 256, 0, stream>>>(w3, W3b, HID, HID, HID);
  phv3_cvt<<<(VOUT * HID + 255) / 256, 256, 0, stream>>>(out_W, Wob, VOUT, HID, HID);

  dim3 g1(NROWS / 128, HID / 64);
  phv3_gemm1<<<g1, 256, 0, stream>>>(F, Wa, in_b, H);
  phv3_gemm2<<<g1, 256, 0, stream>>>(H, W1b, W2b, U);
  phv3_gemm3<<<g1, 256, 0, stream>>>(U, W3b, H, X);
  phv3_ln<<<NROWS / 8, 256, 0, stream>>>(X, ln_g, ln_b, Xn);
  dim3 g4(NROWS / 128, VOUT / 64);
  phv3_gemm4<<<g4, 256, 0, stream>>>(Xn, Wob, out_b, (float*)d_out);
}
